// ContMix_66872640799348
// MI455X (gfx1250) — matrix operand order repaired and checked by reading, not compiled, not run on hardware
//
#include <hip/hip_runtime.h>
#include <hip/hip_bf16.h>
#include <math.h>

// ContMix forward for MI455X (gfx1250, wave32).
// bf16 WMMA (v_wmma_f32_16x16x32_bf16, f32 accumulate) for the three GEMM
// stages; f32 VALU for pool/LN/softmax/neighborhood-AV.
// GEMM main loop: double-buffered LDS + register pipelining + global_prefetch.

typedef __bf16 bf16;
typedef __attribute__((ext_vector_type(16))) __bf16 bf16x16;
typedef __attribute__((ext_vector_type(8)))  __bf16 bf16x8;
typedef __attribute__((ext_vector_type(8)))  float  f32x8;

#define DIMC 256
#define NLOC 4096   // H*W = 64*64
#define NBAT 4

static __device__ inline bf16x16 cat8(bf16x8 lo, bf16x8 hi) {
  bf16x16 r;
#pragma unroll
  for (int i = 0; i < 8; ++i) { r[i] = lo[i]; r[i + 8] = hi[i]; }
  return r;
}

// ---------------- 8x8 average pool: ctx[4,256,56,56] -> kctx[4,256,49] ----
__global__ void pool_kernel(const float* __restrict__ ctx,
                            float* __restrict__ kctx) {
  int o = blockIdx.x * 256 + threadIdx.x;          // 4*256*49 = 50176
  if (o >= NBAT * DIMC * 49) return;
  int l = o % 49;
  int ch = (o / 49) % DIMC;
  int b = o / (49 * DIMC);
  int p = l / 7, q = l % 7;
  const float* src = ctx + (((size_t)b * DIMC + ch) * 56 + p * 8) * 56 + q * 8;
  float s = 0.f;
#pragma unroll
  for (int dy = 0; dy < 8; ++dy)
#pragma unroll
    for (int dx = 0; dx < 8; ++dx) s += src[dy * 56 + dx];
  kctx[o] = s * (1.0f / 64.0f);
}

// ---------------- kf = LN_channel(Wk @ kctx) : [4,256,49] -----------------
__global__ __launch_bounds__(256)
void kgemm_ln_kernel(const float* __restrict__ kctx, const float* __restrict__ Wk,
                     const float* __restrict__ gk, const float* __restrict__ bk,
                     float* __restrict__ kf) {
  const int l = blockIdx.x;        // 0..48
  const int b = blockIdx.y;        // 0..3
  const int o = threadIdx.x;       // output channel
  const float* kc = kctx + (size_t)b * DIMC * 49 + l;
  const float* wr = Wk + (size_t)o * DIMC;
  float acc = 0.f;
  for (int c = 0; c < DIMC; ++c) acc += wr[c] * kc[(size_t)c * 49];
  __shared__ float s1[256], s2[256];
  s1[o] = acc; s2[o] = acc * acc;
  __syncthreads();
  for (int st = 128; st > 0; st >>= 1) {
    if (o < st) { s1[o] += s1[o + st]; s2[o] += s2[o + st]; }
    __syncthreads();
  }
  float mu = s1[0] * (1.f / 256.f);
  float var = s2[0] * (1.f / 256.f) - mu * mu;
  float rstd = rsqrtf(var + 1e-6f);
  kf[((size_t)b * DIMC + o) * 49 + l] = (acc - mu) * rstd * gk[o] + bk[o];
}

// ---------------- PT[b,g,mm,c] = sum_l kf[b,g*64+c,l] * Wproj[moff+mm,l] --
// mm padded to 64 per group (zeros beyond mcnt); stored bf16, row-major in c.
__global__ void pproj_kernel(const float* __restrict__ kf,
                             const float* __restrict__ Wproj,
                             bf16* __restrict__ PT) {
  int idx = blockIdx.x * 256 + threadIdx.x;        // 4*4*64*64 = 65536
  int c = idx & 63;
  int mm = (idx >> 6) & 63;
  int g = (idx >> 12) & 3;
  int b = idx >> 14;
  int mcnt = (g < 2) ? 25 : 49;
  int moff = (g < 2) ? 0 : 25;
  float v = 0.f;
  if (mm < mcnt) {
    const float* kfr = kf + ((size_t)b * DIMC + g * 64 + c) * 49;
    const float* wp = Wproj + (size_t)(moff + mm) * 49;
#pragma unroll
    for (int l = 0; l < 49; ++l) v += kfr[l] * wp[l];
  }
  PT[idx] = (bf16)v;
}

// ---------------- bf16 WMMA GEMM: C[b] = A[256,256] @ B[b][256,4096] ------
// block tile 64(M) x 128(N), K-step 32; 8 waves, wave tile 32x32 (2x2 WMMA).
// Double-buffered LDS: global loads for step k+1 issue before the WMMAs of
// step k; prefetch (global_prefetch_b8) covers step k+2.
template <bool EPI>
__global__ __launch_bounds__(256)
void gemm_kernel(const float* __restrict__ Ag, const float* __restrict__ Bg_all,
                 float* __restrict__ Cg_all, const float* __restrict__ bng,
                 const float* __restrict__ bnb) {
  const int t = threadIdx.x, lane = t & 31, w = t >> 5;
  const int wm = w >> 2, wn = w & 3;
  const int n0 = blockIdx.x * 128, m0 = blockIdx.y * 64;
  const int b = blockIdx.z;
  const float* Bg = Bg_all + (size_t)b * DIMC * NLOC;
  float* Cg = Cg_all + (size_t)b * DIMC * NLOC;

  __shared__ bf16 As[2][64 * 32];       // A tile row-major [m][k]
  __shared__ bf16 Bs[2][128 * 40];      // B tile transposed [n][k], stride 40

  f32x8 acc[2][2];
#pragma unroll
  for (int i = 0; i < 2; ++i)
#pragma unroll
    for (int j = 0; j < 2; ++j)
#pragma unroll
      for (int e = 0; e < 8; ++e) acc[i][j][e] = 0.f;

  // prologue: stage K-step 0 into buffer 0
#pragma unroll
  for (int it = 0; it < 8; ++it) {
    int idx = t + it * 256;
    As[0][idx] = (bf16)Ag[(size_t)(m0 + (idx >> 5)) * DIMC + (idx & 31)];
  }
#pragma unroll
  for (int it = 0; it < 16; ++it) {
    int idx = t + it * 256;
    Bs[0][(idx & 127) * 40 + (idx >> 7)] =
        (bf16)Bg[(size_t)(idx >> 7) * NLOC + n0 + (idx & 127)];
  }
  __syncthreads();

  float aR[8], bR[16];
  for (int ks = 0; ks < 8; ++ks) {
    const int cur = ks & 1;
    if (ks < 7) {                         // issue next tile's global loads
      const int kn = (ks + 1) * 32;
#pragma unroll
      for (int it = 0; it < 8; ++it) {
        int idx = t + it * 256;
        aR[it] = Ag[(size_t)(m0 + (idx >> 5)) * DIMC + kn + (idx & 31)];
      }
#pragma unroll
      for (int it = 0; it < 16; ++it) {
        int idx = t + it * 256;
        bR[it] = Bg[(size_t)(kn + (idx >> 7)) * NLOC + n0 + (idx & 127)];
      }
      if (ks < 6) {                       // pull k+2 toward the WGP
        const int kp = (ks + 2) * 32;
        __builtin_prefetch(&Ag[(size_t)(m0 + (t >> 5)) * DIMC + kp + (t & 31)], 0, 0);
        __builtin_prefetch(&Bg[(size_t)(kp + (t >> 7)) * NLOC + n0 + (t & 127)], 0, 0);
      }
    }

    // compute current buffer
    const int h = lane >> 4;
    bf16x16 afr[2], bfr[2];
#pragma unroll
    for (int i = 0; i < 2; ++i) {
      int row = wm * 32 + i * 16 + (lane & 15);
      const bf16* ap = &As[cur][row * 32 + h * 8];
      afr[i] = cat8(*(const bf16x8*)ap, *(const bf16x8*)(ap + 16));
    }
#pragma unroll
    for (int j = 0; j < 2; ++j) {
      int col = wn * 32 + j * 16 + (lane & 15);
      const bf16* bp = &Bs[cur][col * 40 + h * 8];
      bfr[j] = cat8(*(const bf16x8*)bp, *(const bf16x8*)(bp + 16));
    }
#pragma unroll
    for (int i = 0; i < 2; ++i)
#pragma unroll
      for (int j = 0; j < 2; ++j)
        acc[i][j] = __builtin_amdgcn_wmma_f32_16x16x32_bf16(
            false, afr[i], false, bfr[j], (short)0, acc[i][j], false, false);

    if (ks < 7) {                         // commit next tile to alternate buffer
      const int nxt = cur ^ 1;
#pragma unroll
      for (int it = 0; it < 8; ++it) {
        int idx = t + it * 256;
        As[nxt][idx] = (bf16)aR[it];
      }
#pragma unroll
      for (int it = 0; it < 16; ++it) {
        int idx = t + it * 256;
        Bs[nxt][(idx & 127) * 40 + (idx >> 7)] = (bf16)bR[it];
      }
    }
    __syncthreads();
  }

  const float bnc = rsqrtf(1.0f + 1e-5f);
#pragma unroll
  for (int i = 0; i < 2; ++i)
#pragma unroll
    for (int j = 0; j < 2; ++j)
#pragma unroll
      for (int r = 0; r < 8; ++r) {
        int row = m0 + wm * 32 + i * 16 + (lane >> 4) * 8 + r;
        int col = n0 + wn * 32 + j * 16 + (lane & 15);
        float v = acc[i][j][r];
        if (EPI) v = v * (bng[row] * bnc) + bnb[row];
        Cg[(size_t)row * NLOC + col] = v;
      }
}

// ---------------- LN over channels per pixel; emit bf16 q[b,g,n,c]*SCALE --
__global__ __launch_bounds__(256)
void lnq_kernel(const float* __restrict__ qpre, const float* __restrict__ gq,
                const float* __restrict__ bq, bf16* __restrict__ qn) {
  const int t = threadIdx.x;
  const int loc0 = blockIdx.x * 32, b = blockIdx.y;
  __shared__ float tile[256 * 33];   // [c][loc], stride 33 (bank-padded)
  __shared__ float muS[32], rsS[32];
  const float* src = qpre + (size_t)b * DIMC * NLOC + loc0;
#pragma unroll
  for (int it = 0; it < 32; ++it) {
    int idx = t + it * 256;
    int c = idx >> 5, loc = idx & 31;
    tile[c * 33 + loc] = src[(size_t)c * NLOC + loc];
  }
  __syncthreads();
  if (t < 32) {
    float s = 0.f, ss = 0.f;
    for (int c = 0; c < DIMC; ++c) { float v = tile[c * 33 + t]; s += v; ss += v * v; }
    float mu = s * (1.f / 256.f);
    float var = ss * (1.f / 256.f) - mu * mu;
    muS[t] = mu; rsS[t] = rsqrtf(var + 1e-6f);
  }
  __syncthreads();
  const float scale = 0.125f;                        // HD^-0.5, HD=64
  const int g = t >> 6, c = t & 63;
  const float gv = gq[t], bv = bq[t];
  bf16* dst = qn + (size_t)(b * 4 + g) * NLOC * 64;
  for (int loc = 0; loc < 32; ++loc) {
    float v = tile[t * 33 + loc];
    float qv = ((v - muS[loc]) * rsS[loc] * gv + bv) * scale;
    dst[(size_t)(loc0 + loc) * 64 + c] = (bf16)qv;
  }
}

// ---------------- fused wgt-WMMA + RPB + softmax + neighborhood AV --------
// One block per (b, g, image-row i): 64 pixels.
__global__ __launch_bounds__(256)
void attn_kernel(const bf16* __restrict__ qn, const bf16* __restrict__ PT,
                 const float* __restrict__ x, const float* __restrict__ rpb1,
                 const float* __restrict__ rpb2, float* __restrict__ y) {
  const int t = threadIdx.x, lane = t & 31, w = t >> 5;
  const int i = blockIdx.x, g = blockIdx.y, b = blockIdx.z;
  const int kw = (g < 2) ? 5 : 7;
  const int hk = kw >> 1;
  const int mcnt = kw * kw;
  const int NT = (g < 2) ? 2 : 4;                 // N tiles of 16 (mm padded)
  const int rs = 2 * kw - 1;
  const float* rpbp = (g < 2) ? (rpb1 + g * 81) : (rpb2 + (g - 2) * 169);

  __shared__ __align__(16) char stageRaw[7 * 16 * 64 * 4];  // 28672 B union
  __shared__ float wgtS[64 * 65];                            // [j][mm], padded
  __shared__ float rpbS[169];

  bf16* qS = (bf16*)stageRaw;                // phase 1: q tile [64j][64c]
  bf16* pS = (bf16*)(stageRaw + 8192);       // phase 1: P tile [64mm][64c]
  float* xS = (float*)stageRaw;              // phase 3: x rows [kw][16][64]

  const bf16* qsrc = qn + ((size_t)(b * 4 + g) * NLOC + i * 64) * 64;
  const bf16* psrc = PT + (size_t)(b * 4 + g) * 4096;
#pragma unroll
  for (int it = 0; it < 16; ++it) {
    int idx = t + it * 256;
    qS[idx] = qsrc[idx];
    pS[idx] = psrc[idx];
  }
  if (t < rs * rs) rpbS[t] = rpbp[t];
  __syncthreads();

  { // wgt[j,mm] = sum_c q[j,c] * PT[mm,c] via WMMA (K=64 -> 2 steps)
    const int mt = w >> 1;
    const int h = lane >> 4;
    for (int s = 0; s < (NT >> 1); ++s) {
      int nt = (w & 1) + 2 * s;
      f32x8 accv;
#pragma unroll
      for (int e = 0; e < 8; ++e) accv[e] = 0.f;
#pragma unroll
      for (int ks = 0; ks < 2; ++ks) {
        int row = mt * 16 + (lane & 15);
        const bf16* ap = qS + row * 64 + ks * 32 + h * 8;
        bf16x16 a = cat8(*(const bf16x8*)ap, *(const bf16x8*)(ap + 16));
        int col = nt * 16 + (lane & 15);
        const bf16* bp = pS + col * 64 + ks * 32 + h * 8;
        bf16x16 bb = cat8(*(const bf16x8*)bp, *(const bf16x8*)(bp + 16));
        accv = __builtin_amdgcn_wmma_f32_16x16x32_bf16(
            false, a, false, bb, (short)0, accv, false, false);
      }
#pragma unroll
      for (int r = 0; r < 8; ++r) {
        int jr = mt * 16 + h * 8 + r;
        int mm = nt * 16 + (lane & 15);
        wgtS[jr * 65 + mm] = accv[r];
      }
    }
  }
  __syncthreads();

  const int si = min(max(i - hk, 0), 64 - kw);
  if (t < 64) {                                   // RPB + softmax, one j/thread
    int j = t;
    int sjl = min(max(j - hk, 0), 64 - kw);
    int bi = i - si, bj = j - sjl;
    float mx = -1e30f;
    for (int mm = 0; mm < mcnt; ++mm) {
      int p = mm / kw, q = mm - p * kw;
      float v = wgtS[j * 65 + mm] + rpbS[(kw - 1 - bi + p) * rs + (kw - 1 - bj + q)];
      wgtS[j * 65 + mm] = v;
      mx = fmaxf(mx, v);
    }
    float sum = 0.f;
    for (int mm = 0; mm < mcnt; ++mm) {
      float e = expf(wgtS[j * 65 + mm] - mx);
      wgtS[j * 65 + mm] = e;
      sum += e;
    }
    float inv = 1.f / sum;
    for (int mm = 0; mm < mcnt; ++mm) wgtS[j * 65 + mm] *= inv;
  }

  // AV: out[j,c] = sum_{p,q} attn[j,p*kw+q] * x[b, g*64+c, si+p, sj+q]
  const int j = t & 63, ch0 = t >> 6;
  const int sj = min(max(j - hk, 0), 64 - kw);
  const float* xg = x + ((size_t)b * DIMC + g * 64) * NLOC;
  for (int chunk = 0; chunk < 4; ++chunk) {       // 16 channels per chunk
    __syncthreads();                              // xS free / prev chunk done
    int tot = kw * 16 * 64;
    for (int idx = t; idx < tot; idx += 256) {
      int p = idx / 1024;
      int rem = idx - p * 1024;
      int ch = rem >> 6, col = rem & 63;
      xS[idx] = xg[(size_t)(chunk * 16 + ch) * NLOC + (si + p) * 64 + col];
    }
    __syncthreads();
#pragma unroll
    for (int u = 0; u < 4; ++u) {
      int ch = ch0 + 4 * u;
      float a2 = 0.f;
      for (int p = 0; p < kw; ++p) {
        const float* xr = xS + (p * 16 + ch) * 64 + sj;
        const float* ar = wgtS + j * 65 + p * kw;
        for (int q = 0; q < kw; ++q) a2 += ar[q] * xr[q];
      }
      y[((size_t)b * DIMC + g * 64 + chunk * 16 + ch) * NLOC + i * 64 + j] = a2;
    }
  }
}

// --------------------------------------------------------------------------
extern "C" void kernel_launch(void* const* d_in, const int* in_sizes, int n_in,
                              void* d_out, int out_size, void* d_ws, size_t ws_size,
                              hipStream_t stream) {
  (void)in_sizes; (void)n_in; (void)out_size; (void)ws_size;
  const float* x     = (const float*)d_in[0];
  const float* ctx   = (const float*)d_in[1];
  const float* Wq    = (const float*)d_in[2];
  const float* gq    = (const float*)d_in[3];
  const float* bq    = (const float*)d_in[4];
  const float* Wk    = (const float*)d_in[5];
  const float* gk    = (const float*)d_in[6];
  const float* bk    = (const float*)d_in[7];
  const float* Wproj = (const float*)d_in[8];
  const float* rpb1  = (const float*)d_in[9];
  const float* rpb2  = (const float*)d_in[10];
  const float* Wdy   = (const float*)d_in[11];
  const float* bng   = (const float*)d_in[12];
  const float* bnb   = (const float*)d_in[13];
  float* out = (float*)d_out;

  char* ws = (char*)d_ws;
  float* qpre_y = (float*)ws;                          // 16 MiB (q_pre, then y)
  bf16* qn      = (bf16*)(ws + 16777216);              // 8 MiB
  float* kctx   = (float*)(ws + 16777216 + 8388608);   // 200704 B
  float* kf     = (float*)(ws + 16777216 + 8388608 + 200704);
  bf16* PT      = (bf16*)(ws + 16777216 + 8388608 + 401408);

  pool_kernel<<<196, 256, 0, stream>>>(ctx, kctx);
  kgemm_ln_kernel<<<dim3(49, 4), 256, 0, stream>>>(kctx, Wk, gk, bk, kf);
  pproj_kernel<<<256, 256, 0, stream>>>(kf, Wproj, PT);
  gemm_kernel<false><<<dim3(32, 4, 4), 256, 0, stream>>>(Wq, x, qpre_y, nullptr, nullptr);
  lnq_kernel<<<dim3(128, 4), 256, 0, stream>>>(qpre_y, gq, bq, qn);
  attn_kernel<<<dim3(64, 4, 4), 256, 0, stream>>>(qn, PT, x, rpb1, rpb2, qpre_y);
  gemm_kernel<true><<<dim3(32, 4, 4), 256, 0, stream>>>(Wdy, qpre_y, out, bng, bnb);
}
